// MambaEncoder_40295383171086
// MI455X (gfx1250) — matrix operand order repaired and checked by reading, not compiled, not run on hardware
//
#include <hip/hip_runtime.h>
#include <hip/hip_bf16.h>

// ---------------- problem constants ----------------
#define BATCH   4
#define SEQ     4096
#define BLROWS  (BATCH * SEQ)      // 16384
#define DM      128
#define DI      256
#define NSTATE  16
#define KCONV   4
#define DTRANK  8
#define XPROJ_N 40                 // DTRANK + 2*NSTATE
#define XPROJ_NP 48                // zero-padded to whole 16-col tiles
#define DBL_LD  48                 // padded leading dim for dbl
#define NLAYERS 4
#define NL      (2 * NLAYERS)
#define NCHUNK  64
#define LCHUNK  (SEQ / NCHUNK)     // 64

typedef __attribute__((ext_vector_type(16))) __bf16 v16bf;
typedef __attribute__((ext_vector_type(8)))  __bf16 v8bf;
typedef __attribute__((ext_vector_type(8)))  float  v8f;

// ---------------- helpers ----------------
__device__ __forceinline__ __bf16 f2bf(float f) {
  unsigned u = __float_as_uint(f);
  u += 0x7FFFu + ((u >> 16) & 1u);          // round-to-nearest-even
  unsigned short h = (unsigned short)(u >> 16);
  return __builtin_bit_cast(__bf16, h);
}
__device__ __forceinline__ float sigmoidf_(float x) { return 1.f / (1.f + __expf(-x)); }
__device__ __forceinline__ float siluf_(float x)    { return x * sigmoidf_(x); }
__device__ __forceinline__ float softplusf_(float x){ return (x > 20.f) ? x : __logf(1.f + __expf(x)); }

// ---------------- fp32 -> bf16 pack (4 elements / thread) ----------------------
__global__ void pack_bf16_kernel(const float* __restrict__ src,
                                 __bf16* __restrict__ dst, int n4)
{
  int i = blockIdx.x * blockDim.x + threadIdx.x;
  if (i >= n4) return;
  float4 v = ((const float4*)src)[i];
  dst[4 * i + 0] = f2bf(v.x);
  dst[4 * i + 1] = f2bf(v.y);
  dst[4 * i + 2] = f2bf(v.z);
  dst[4 * i + 3] = f2bf(v.w);
}

// pack xproj weights (NL, 40, 256) -> (NL, 48, 256) bf16 with zero pad rows
__global__ void pack_xproj_kernel(const float* __restrict__ src,
                                  __bf16* __restrict__ dst)
{
  int idx = blockIdx.x * blockDim.x + threadIdx.x;      // NL*48*256
  int k   = idx & (DI - 1);
  int row = (idx >> 8) % XPROJ_NP;
  int lw  = (idx >> 8) / XPROJ_NP;
  float v = (row < XPROJ_N) ? src[((size_t)lw * XPROJ_N + row) * DI + k] : 0.f;
  dst[idx] = f2bf(v);
}

// ---------------- bf16 WMMA GEMM:  C = A(MxK) * W(NxK)^T  (+ residual) ---------
// Guard-free: N padded to 16-col tiles, operands pre-packed bf16.
// Each wave computes a 32x16 tile: one B fragment feeds two WMMAs.
// Layouts per CDNA5 ISA 7.12.2:
//   A 16x32 bf16 : lane(0..15)=row M; e<8 -> K=hi*8+e ; e>=8 -> K=16+hi*8+(e-8)
//   B 32x16 bf16 : lane&15 = column N, K order same as A
//   C/D f32      : lane&15 = N, vgpr r -> M = hi*8 + r
template<bool LOAD_C, bool STORE_BF, int K>
__global__ void gemm_bf16_wmma(const __bf16* __restrict__ A, int lda,
                               const __bf16* __restrict__ W, int ldw,
                               const float* __restrict__ Cin,
                               float* __restrict__ Cout,
                               __bf16* __restrict__ Cbf,
                               int ldc, int Ntiles)
{
  const int lane  = threadIdx.x & 31;
  const int wave  = threadIdx.x >> 5;
  const int ntile = blockIdx.x * 4 + wave;
  if (ntile >= Ntiles) return;            // wave-uniform: EXEC all-ones for WMMA

  const int m0  = blockIdx.y << 5;        // 32 rows per wave
  const int n0  = ntile << 4;
  const int r   = lane & 15;
  const int hi  = lane >> 4;
  const int col = n0 + r;

  v8f acc0, acc1;
#pragma unroll
  for (int i = 0; i < 8; ++i) {
    acc0[i] = LOAD_C ? Cin[(size_t)(m0 + hi * 8 + i) * ldc + col] : 0.f;
    acc1[i] = LOAD_C ? Cin[(size_t)(m0 + 16 + hi * 8 + i) * ldc + col] : 0.f;
  }

  const __bf16* arow0 = A + (size_t)(m0 + r) * lda;
  const __bf16* arow1 = A + (size_t)(m0 + 16 + r) * lda;
  const __bf16* wrow  = W + (size_t)col * ldw;

#pragma unroll
  for (int k0 = 0; k0 < K; k0 += 32) {
    __builtin_prefetch(arow0 + k0 + 256, 0, 3);    // global_prefetch_b8
    __builtin_prefetch(arow1 + k0 + 256, 0, 3);
    v8bf a00 = *(const v8bf*)(arow0 + k0 + hi * 8);
    v8bf a01 = *(const v8bf*)(arow0 + k0 + 16 + hi * 8);
    v16bf af0 = __builtin_shufflevector(a00, a01, 0, 1, 2, 3, 4, 5, 6, 7,
                                        8, 9, 10, 11, 12, 13, 14, 15);
    v8bf a10 = *(const v8bf*)(arow1 + k0 + hi * 8);
    v8bf a11 = *(const v8bf*)(arow1 + k0 + 16 + hi * 8);
    v16bf af1 = __builtin_shufflevector(a10, a11, 0, 1, 2, 3, 4, 5, 6, 7,
                                        8, 9, 10, 11, 12, 13, 14, 15);
    v8bf b0 = *(const v8bf*)(wrow + k0 + hi * 8);
    v8bf b1 = *(const v8bf*)(wrow + k0 + 16 + hi * 8);
    v16bf bv = __builtin_shufflevector(b0, b1, 0, 1, 2, 3, 4, 5, 6, 7,
                                       8, 9, 10, 11, 12, 13, 14, 15);
    acc0 = __builtin_amdgcn_wmma_f32_16x16x32_bf16(false, af0, false, bv,
                                                   (short)0, acc0, false, false);
    acc1 = __builtin_amdgcn_wmma_f32_16x16x32_bf16(false, af1, false, bv,
                                                   (short)0, acc1, false, false);
  }

#pragma unroll
  for (int i = 0; i < 8; ++i) {
    size_t o0 = (size_t)(m0 + hi * 8 + i) * ldc + col;
    size_t o1 = (size_t)(m0 + 16 + hi * 8 + i) * ldc + col;
    Cout[o0] = acc0[i];
    Cout[o1] = acc1[i];
    if (STORE_BF) { Cbf[o0] = f2bf(acc0[i]); Cbf[o1] = f2bf(acc1[i]); }
  }
}

// ---------------- embed: h = x @ ew^T + eb ; fwd + time-reversed bwd (+bf16) ---
__global__ void embed_kernel(const float* __restrict__ x,
                             const float* __restrict__ ew,
                             const float* __restrict__ eb,
                             float* __restrict__ hf, float* __restrict__ hb,
                             __bf16* __restrict__ hf_bf, __bf16* __restrict__ hb_bf)
{
  int idx = blockIdx.x * blockDim.x + threadIdx.x;      // BLROWS*DM
  int d  = idx & (DM - 1);
  int bl = idx >> 7;
  int l  = bl & (SEQ - 1);
  int b  = bl >> 12;
  float s = eb[d];
#pragma unroll
  for (int i = 0; i < 5; ++i) s += x[(size_t)bl * 5 + i] * ew[d * 5 + i];
  size_t j = (size_t)(b * SEQ + (SEQ - 1 - l)) * DM + d;
  hf[idx] = s;           hb[j] = s;
  hf_bf[idx] = f2bf(s);  hb_bf[j] = f2bf(s);
}

// ---------------- causal depthwise conv (K=4) + bias + SiLU (+bf16) ------------
__global__ void conv_silu_kernel(const float* __restrict__ xz,
                                 const float* __restrict__ cw,
                                 const float* __restrict__ cb,
                                 float* __restrict__ xc,
                                 __bf16* __restrict__ xc_bf)
{
  int idx = blockIdx.x * blockDim.x + threadIdx.x;      // BLROWS*DI
  int ch = idx & (DI - 1);
  int bl = idx >> 8;
  int l  = bl & (SEQ - 1);
  int b  = bl >> 12;
  float s = cb[ch];
#pragma unroll
  for (int k = 0; k < KCONV; ++k) {
    int ls = l + k - (KCONV - 1);
    float v = (ls >= 0) ? xz[(size_t)(b * SEQ + ls) * (2 * DI) + ch] : 0.f;
    s += cw[ch * KCONV + k] * v;
  }
  float o = siluf_(s);
  xc[idx] = o;
  xc_bf[idx] = f2bf(o);
}

// ---------------- delta = softplus(dtr @ dw^T + db) ----------------------------
__global__ void delta_kernel(const float* __restrict__ dbl,
                             const float* __restrict__ dw,
                             const float* __restrict__ db,
                             float* __restrict__ delta_)
{
  int idx = blockIdx.x * blockDim.x + threadIdx.x;      // BLROWS*DI
  int d  = idx & (DI - 1);
  int bl = idx >> 8;
  const float* dtr = dbl + (size_t)bl * DBL_LD;
  float s = db[d];
#pragma unroll
  for (int rr = 0; rr < DTRANK; ++rr) s += dtr[rr] * dw[d * DTRANK + rr];
  delta_[idx] = softplusf_(s);
}

// ---------------- scan pass 1: per-chunk local scan -> (prod dA, h_end) -------
__global__ void scan_p1(const float* __restrict__ delta_,
                        const float* __restrict__ xc,
                        const float* __restrict__ dbl,
                        const float* __restrict__ Alog,
                        float* __restrict__ aprod,
                        float* __restrict__ hend)
{
  int idx = blockIdx.x * blockDim.x + threadIdx.x;      // BATCH*NCHUNK*DI
  int d  = idx & (DI - 1);
  int bc = idx >> 8;
  int c  = bc & (NCHUNK - 1);
  int b  = bc >> 6;

  float An[NSTATE], ap[NSTATE], h[NSTATE];
#pragma unroll
  for (int n = 0; n < NSTATE; ++n) {
    An[n] = -__expf(Alog[d * NSTATE + n]);
    ap[n] = 1.f; h[n] = 0.f;
  }
  int base_l = c * LCHUNK;
  for (int t = 0; t < LCHUNK; ++t) {
    int bl = b * SEQ + base_l + t;
    float dl = delta_[(size_t)bl * DI + d];
    float xv = xc[(size_t)bl * DI + d];
    const float* Bm = dbl + (size_t)bl * DBL_LD + DTRANK;
    float dx = dl * xv;
#pragma unroll
    for (int n = 0; n < NSTATE; ++n) {
      float dA = __expf(dl * An[n]);
      h[n]  = dA * h[n] + Bm[n] * dx;
      ap[n] *= dA;
    }
  }
  size_t o = (size_t)idx * NSTATE;
#pragma unroll
  for (int n = 0; n < NSTATE; ++n) { aprod[o + n] = ap[n]; hend[o + n] = h[n]; }
}

// ---------------- scan pass 2: sequential carry combine across chunks ---------
__global__ void scan_p2(const float* __restrict__ aprod,
                        const float* __restrict__ hend,
                        float* __restrict__ prefix)
{
  int idx = blockIdx.x * blockDim.x + threadIdx.x;      // BATCH*DI
  int d = idx & (DI - 1);
  int b = idx >> 8;
  float s[NSTATE];
#pragma unroll
  for (int n = 0; n < NSTATE; ++n) s[n] = 0.f;
  for (int c = 0; c < NCHUNK; ++c) {
    size_t o = ((size_t)(b * NCHUNK + c) * DI + d) * NSTATE;
#pragma unroll
    for (int n = 0; n < NSTATE; ++n) prefix[o + n] = s[n];
#pragma unroll
    for (int n = 0; n < NSTATE; ++n) s[n] = aprod[o + n] * s[n] + hend[o + n];
  }
}

// ---------------- scan pass 3: replay with prefix; fuse y, D-skip, SiLU gate --
// Emits yact directly in bf16 (only consumer is the outproj WMMA GEMM).
__global__ void scan_p3(const float* __restrict__ delta_,
                        const float* __restrict__ xc,
                        const float* __restrict__ dbl,
                        const float* __restrict__ Alog,
                        const float* __restrict__ Dsk,
                        const float* __restrict__ xz,
                        const float* __restrict__ prefix,
                        __bf16* __restrict__ yact_bf)
{
  int idx = blockIdx.x * blockDim.x + threadIdx.x;      // BATCH*NCHUNK*DI
  int d  = idx & (DI - 1);
  int bc = idx >> 8;
  int c  = bc & (NCHUNK - 1);
  int b  = bc >> 6;

  float An[NSTATE], s[NSTATE];
  size_t o = (size_t)idx * NSTATE;
#pragma unroll
  for (int n = 0; n < NSTATE; ++n) {
    An[n] = -__expf(Alog[d * NSTATE + n]);
    s[n]  = prefix[o + n];
  }
  float Dv = Dsk[d];
  int base_l = c * LCHUNK;
  for (int t = 0; t < LCHUNK; ++t) {
    int bl = b * SEQ + base_l + t;
    float dl = delta_[(size_t)bl * DI + d];
    float xv = xc[(size_t)bl * DI + d];
    const float* Bm = dbl + (size_t)bl * DBL_LD + DTRANK;
    const float* Cm = Bm + NSTATE;
    float dx = dl * xv;
    float y = 0.f;
#pragma unroll
    for (int n = 0; n < NSTATE; ++n) {
      float dA = __expf(dl * An[n]);
      s[n] = dA * s[n] + Bm[n] * dx;
      y += s[n] * Cm[n];
    }
    y += Dv * xv;
    float z = xz[(size_t)bl * (2 * DI) + DI + d];
    yact_bf[(size_t)bl * DI + d] = f2bf(y * siluf_(z));
  }
}

// ---------------- final: concat(fwd, flipped bwd) + LayerNorm -----------------
__global__ void ln_kernel(const float* __restrict__ hf,
                          const float* __restrict__ hb,
                          const float* __restrict__ g,
                          const float* __restrict__ be,
                          float* __restrict__ out)
{
  int bl = blockIdx.x;
  int ch = threadIdx.x;                                  // 256
  int l  = bl & (SEQ - 1);
  int b  = bl >> 12;
  float v = (ch < DM) ? hf[(size_t)bl * DM + ch]
                      : hb[(size_t)(b * SEQ + (SEQ - 1 - l)) * DM + (ch - DM)];
  __shared__ float red[256];
  red[ch] = v; __syncthreads();
  for (int st = 128; st > 0; st >>= 1) { if (ch < st) red[ch] += red[ch + st]; __syncthreads(); }
  float mu = red[0] * (1.f / 256.f); __syncthreads();
  float dv = v - mu;
  red[ch] = dv * dv; __syncthreads();
  for (int st = 128; st > 0; st >>= 1) { if (ch < st) red[ch] += red[ch + st]; __syncthreads(); }
  float var = red[0] * (1.f / 256.f);
  out[(size_t)bl * 256 + ch] = dv * rsqrtf(var + 1e-5f) * g[ch] + be[ch];
}

// ---------------- host launcher ----------------------------------------------
extern "C" void kernel_launch(void* const* d_in, const int* in_sizes, int n_in,
                              void* d_out, int out_size, void* d_ws, size_t ws_size,
                              hipStream_t stream) {
  (void)in_sizes; (void)n_in; (void)out_size; (void)ws_size;
  const float* x        = (const float*)d_in[0];
  const float* embed_w  = (const float*)d_in[1];
  const float* embed_b  = (const float*)d_in[2];
  const float* inproj_w = (const float*)d_in[3];
  const float* conv_w   = (const float*)d_in[4];
  const float* conv_b   = (const float*)d_in[5];
  const float* xproj_w  = (const float*)d_in[6];
  const float* dtproj_w = (const float*)d_in[7];
  const float* dtproj_b = (const float*)d_in[8];
  const float* A_log    = (const float*)d_in[9];
  const float* D_skip   = (const float*)d_in[10];
  const float* outproj_w= (const float*)d_in[11];
  const float* ln_g     = (const float*)d_in[12];
  const float* ln_b     = (const float*)d_in[13];
  float* out = (float*)d_out;

  // ---- workspace carving: fp32 region then bf16 region ----
  float* ws = (float*)d_ws;
  size_t off = 0;
  float* h_fwd  = ws + off; off += (size_t)BLROWS * DM;
  float* h_bwd  = ws + off; off += (size_t)BLROWS * DM;
  float* xz     = ws + off; off += (size_t)BLROWS * 2 * DI;
  float* xc     = ws + off; off += (size_t)BLROWS * DI;
  float* dbl    = ws + off; off += (size_t)BLROWS * DBL_LD;
  float* delta_ = ws + off; off += (size_t)BLROWS * DI;
  float* aprod  = ws + off; off += (size_t)BATCH * NCHUNK * DI * NSTATE;
  float* hend   = ws + off; off += (size_t)BATCH * NCHUNK * DI * NSTATE;
  float* prefix = ws + off; off += (size_t)BATCH * NCHUNK * DI * NSTATE;

  __bf16* bws = (__bf16*)(ws + off);
  size_t boff = 0;
  __bf16* hf_bf   = bws + boff; boff += (size_t)BLROWS * DM;
  __bf16* hb_bf   = bws + boff; boff += (size_t)BLROWS * DM;
  __bf16* xc_bf   = bws + boff; boff += (size_t)BLROWS * DI;
  __bf16* yact_bf = bws + boff; boff += (size_t)BLROWS * DI;
  __bf16* w_in_bf = bws + boff; boff += (size_t)NL * 2 * DI * DM;
  __bf16* w_xp_bf = bws + boff; boff += (size_t)NL * XPROJ_NP * DI;
  __bf16* w_out_bf= bws + boff; boff += (size_t)NL * DM * DI;

  // ---- pack all weights to bf16 once per launch (xproj zero-padded to 48) ----
  {
    int n4a = NL * 2 * DI * DM / 4;
    int n4c = NL * DM * DI / 4;
    int nxp = NL * XPROJ_NP * DI;
    pack_bf16_kernel<<<(n4a + 255) / 256, 256, 0, stream>>>(inproj_w,  w_in_bf,  n4a);
    pack_bf16_kernel<<<(n4c + 255) / 256, 256, 0, stream>>>(outproj_w, w_out_bf, n4c);
    pack_xproj_kernel<<<(nxp + 255) / 256, 256, 0, stream>>>(xproj_w, w_xp_bf);
  }

  embed_kernel<<<(BLROWS * DM) / 256, 256, 0, stream>>>(x, embed_w, embed_b,
                                                        h_fwd, h_bwd, hf_bf, hb_bf);

  for (int dir = 0; dir < 2; ++dir) {
    float*  h    = dir ? h_bwd : h_fwd;
    __bf16* h_bf = dir ? hb_bf : hf_bf;
    for (int li = 0; li < NLAYERS; ++li) {
      int Lw = dir * NLAYERS + li;
      const float*  cw = conv_w   + (size_t)Lw * DI * KCONV;
      const float*  cb = conv_b   + (size_t)Lw * DI;
      const float*  dw = dtproj_w + (size_t)Lw * DI * DTRANK;
      const float*  db = dtproj_b + (size_t)Lw * DI;
      const float*  Al = A_log    + (size_t)Lw * DI * NSTATE;
      const float*  Dk = D_skip   + (size_t)Lw * DI;
      const __bf16* iw = w_in_bf  + (size_t)Lw * 2 * DI * DM;
      const __bf16* xw = w_xp_bf  + (size_t)Lw * XPROJ_NP * DI;
      const __bf16* ow = w_out_bf + (size_t)Lw * DM * DI;

      // inproj: xz = h @ iw^T   (M=16384, N=512, K=128)
      dim3 g1((2 * DI / 16) / 4, BLROWS / 32);
      gemm_bf16_wmma<false, false, DM><<<g1, 128, 0, stream>>>(
          h_bf, DM, iw, DM, nullptr, xz, nullptr, 2 * DI, 2 * DI / 16);
      conv_silu_kernel<<<(BLROWS * DI) / 256, 256, 0, stream>>>(xz, cw, cb, xc, xc_bf);

      // xproj: dbl = xc @ xw^T  (M=16384, N=48 padded (3 tiles), K=256)
      dim3 g2(1, BLROWS / 32);
      gemm_bf16_wmma<false, false, DI><<<g2, 128, 0, stream>>>(
          xc_bf, DI, xw, DI, nullptr, dbl, nullptr, DBL_LD, XPROJ_NP / 16);
      delta_kernel<<<(BLROWS * DI) / 256, 256, 0, stream>>>(dbl, dw, db, delta_);

      scan_p1<<<(BATCH * NCHUNK * DI) / 256, 256, 0, stream>>>(delta_, xc, dbl, Al, aprod, hend);
      scan_p2<<<(BATCH * DI) / 256, 256, 0, stream>>>(aprod, hend, prefix);
      scan_p3<<<(BATCH * NCHUNK * DI) / 256, 256, 0, stream>>>(delta_, xc, dbl, Al, Dk,
                                                               xz, prefix, yact_bf);

      // outproj + residual: h += yact @ ow^T ; also refresh h_bf for next layer
      dim3 g3((DM / 16) / 4, BLROWS / 32);
      gemm_bf16_wmma<true, true, DI><<<g3, 128, 0, stream>>>(
          yact_bf, DI, ow, DI, h, h, h_bf, DM, DM / 16);
    }
  }
  ln_kernel<<<BLROWS, 256, 0, stream>>>(h_fwd, h_bwd, ln_g, ln_b, out);
}
